// TipAdapterFLinear_90847148245578
// MI455X (gfx1250) — matrix operand order repaired and checked by reading, not compiled, not run on hardware
//
#include <hip/hip_runtime.h>
#include <hip/hip_bf16.h>
#include <stdint.h>

// Problem constants (from reference)
#define B_ROWS 2048
#define D_IN   1024
#define D_OUT  512
#define N_CACHE 50000
#define NPAD   50048              // pad cache dim to multiple of 128
#define NSPLIT 4                  // split-K over cache dim for occupancy
#define NTILES (NPAD / 128)       // 391
#define TILES_PER_SPLIT ((NTILES + NSPLIT - 1) / NSPLIT)  // 98

typedef __bf16        bf16x16 __attribute__((ext_vector_type(16)));
typedef float         f32x8   __attribute__((ext_vector_type(8)));
typedef unsigned int  u32x4   __attribute__((ext_vector_type(4)));

union FragU { u32x4 u[2]; bf16x16 v; };

__device__ __forceinline__ unsigned short f2bf(float f) {
  unsigned int u = __float_as_uint(f);
  u += 0x7FFFu + ((u >> 16) & 1u);   // round-to-nearest-even
  return (unsigned short)(u >> 16);
}

// A-fragment style: two 16B chunks via separate pointers (adjacent at every call site)
__device__ __forceinline__ bf16x16 frag_split(const unsigned short* p0, const unsigned short* p1) {
  FragU f;
  f.u[0] = *(const u32x4*)p0;
  f.u[1] = *(const u32x4*)p1;
  return f.v;
}
// B-fragment style: contiguous 32B
__device__ __forceinline__ bf16x16 frag_contig(const unsigned short* p) {
  FragU f;
  f.u[0] = *(const u32x4*)p;
  f.u[1] = *(const u32x4*)(p + 8);
  return f.v;
}

__device__ __forceinline__ f32x8 wmma_bf16(bf16x16 a, bf16x16 b, f32x8 c) {
  return __builtin_amdgcn_wmma_f32_16x16x32_bf16(false, a, false, b, (short)0, c, false, false);
}

// ---------------------------------------------------------------- conversions
__global__ void __launch_bounds__(256) k_cvt(const float* __restrict__ src,
                                             unsigned short* __restrict__ dst, int n) {
  int i = blockIdx.x * 256 + threadIdx.x;
  if (i < n) dst[i] = f2bf(src[i]);
}

__global__ void __launch_bounds__(256) k_cvt_pad_rows(const float* __restrict__ src,
                                                      unsigned short* __restrict__ dst,
                                                      int rows_valid, int rows_pad, int cols) {
  int i = blockIdx.x * 256 + threadIdx.x;
  int n = rows_pad * cols;
  if (i >= n) return;
  int r = i / cols;
  dst[i] = (r < rows_valid) ? f2bf(src[i]) : (unsigned short)0;
}

// src [N_CACHE][D_OUT] f32  ->  dst [D_OUT][NPAD] bf16 (transposed, zero-padded)
__global__ void __launch_bounds__(256) k_transpose_ct(const float* __restrict__ src,
                                                      unsigned short* __restrict__ dst) {
  __shared__ float t[32][33];
  int tx = threadIdx.x;       // 0..31
  int ty = threadIdx.y;       // 0..7
  int c0 = blockIdx.x * 32;   // cache tile base
  int d0 = blockIdx.y * 32;   // dout tile base
#pragma unroll
  for (int j = 0; j < 4; ++j) {
    int cache = c0 + ty + 8 * j;
    float v = 0.0f;
    if (cache < N_CACHE) v = src[(size_t)cache * D_OUT + d0 + tx];
    t[ty + 8 * j][tx] = v;
  }
  __syncthreads();
#pragma unroll
  for (int j = 0; j < 4; ++j) {
    int dout  = d0 + ty + 8 * j;
    int cache = c0 + tx;
    dst[(size_t)dout * NPAD + cache] = f2bf(t[tx][ty + 8 * j]);
  }
}

// ---------------------------------------------------------------- projection
// proj = x @ W^T + b   (one 16x16 output tile per wave, K = 1024)
__global__ void __launch_bounds__(256) k_proj(const unsigned short* __restrict__ Xb,
                                              const unsigned short* __restrict__ Wb,
                                              const float* __restrict__ bias,
                                              float* __restrict__ proj,
                                              unsigned short* __restrict__ Pb) {
  const int lane = threadIdx.x & 31;
  const int w    = threadIdx.x >> 5;
  const int ln   = lane & 15;
  const int h    = lane >> 4;
  const int t  = blockIdx.x * 8 + w;   // 4096 tiles total
  const int mi = t >> 5;               // 0..127 (batch tile)
  const int ni = t & 31;               // 0..31  (dout tile)

  const int arow = mi * 16 + ln;
  const int bcol = ni * 16 + ln;
  const unsigned short* ap = Xb + (size_t)arow * D_IN + 16 * h;  // A: contiguous 16-K per lane
  const unsigned short* bp = Wb + (size_t)bcol * D_IN + 16 * h;  // B: contiguous 16-K per lane

  f32x8 c = {};
#pragma unroll
  for (int f = 0; f < D_IN / 32; ++f) {
    bf16x16 a = frag_split(ap + 32 * f, ap + 32 * f + 8);
    bf16x16 b = frag_contig(bp + 32 * f);
    c = wmma_bf16(a, b, c);
  }
  float bb = bias[bcol];
#pragma unroll
  for (int v = 0; v < 8; ++v) {
    int row = mi * 16 + v + 8 * h;     // C/D: M = v + 8*half, N = ln
    int col = ni * 16 + ln;
    float val = c[v] + bb;
    proj[(size_t)row * D_OUT + col] = val;
    Pb[(size_t)row * D_OUT + col]   = f2bf(val);
  }
}

// ---------------------------------------------------------------- fused flash retrieval
// Grid: 128 batch-tiles x NSPLIT cache splits; block = 8 waves.
// Per WG: 16 batch rows; per wave: 16x16 S tile + 64 D_OUT columns of acc.
__global__ void __launch_bounds__(256) k_flash(const unsigned short* __restrict__ Pb,
                                               const unsigned short* __restrict__ CIb,  // [NPAD][512]
                                               const unsigned short* __restrict__ CTt,  // [512][NPAD]
                                               const float* __restrict__ beta_p,
                                               float* __restrict__ m_part,
                                               float* __restrict__ l_part,
                                               float* __restrict__ acc_part) {
  __shared__ unsigned short plds[16 * 128];  // p tile (bf16), row-major [row][cachecol]
  __shared__ float red_max[8][16];
  __shared__ float red_sum[8][16];

  const int lane  = threadIdx.x & 31;
  const int w     = threadIdx.x >> 5;
  const int ln    = lane & 15;
  const int h     = lane >> 4;
  const int bi    = blockIdx.x & 127;
  const int split = blockIdx.x >> 7;
  const float beta = beta_p[0];

  // Resident A fragments of the projected tile (16 rows x K=512) — 128 VGPRs
  const int arow = bi * 16 + ln;
  const unsigned short* prow = Pb + (size_t)arow * D_OUT + 16 * h;
  bf16x16 afrag[16];
#pragma unroll
  for (int f = 0; f < 16; ++f)
    afrag[f] = frag_split(prow + 32 * f, prow + 32 * f + 8);

  float m_run[8], l_run[8];
  f32x8 acc[4];
  const f32x8 zero = {};
#pragma unroll
  for (int v = 0; v < 8; ++v) { m_run[v] = -1e30f; l_run[v] = 0.0f; }
#pragma unroll
  for (int n = 0; n < 4; ++n) acc[n] = zero;

  const int t0 = split * TILES_PER_SPLIT;
  int t1 = t0 + TILES_PER_SPLIT; if (t1 > NTILES) t1 = NTILES;

  for (int tile = t0; tile < t1; ++tile) {
    const int nb   = tile * 128;
    const int colc = nb + 16 * w + ln;        // cache row for this lane's S column

    // ---- S tile: P[16x512] @ CI[col,:]^T  (16 wmma, K=512)
    const unsigned short* cip = CIb + (size_t)colc * D_OUT + 16 * h;
    f32x8 s = {};
#pragma unroll
    for (int f = 0; f < 16; ++f) {
      bf16x16 b = frag_contig(cip + 32 * f);
      s = wmma_bf16(afrag[f], b, s);
    }
    if (colc + 128 < NPAD)
      __builtin_prefetch(CIb + (size_t)(colc + 128) * D_OUT + 16 * h, 0, 1);

    float sg[8];
#pragma unroll
    for (int v = 0; v < 8; ++v) {
      float sv = beta * s[v];
      if (colc >= N_CACHE) sv = -1e30f;       // mask padded cache rows
      sg[v] = sv;
    }

    // ---- row max over this wave's 16 columns (cross-lane within 16-lane half)
    float wmax[8];
#pragma unroll
    for (int v = 0; v < 8; ++v) {
      float r = sg[v];
      r = fmaxf(r, __shfl_xor(r, 1, 32));
      r = fmaxf(r, __shfl_xor(r, 2, 32));
      r = fmaxf(r, __shfl_xor(r, 4, 32));
      r = fmaxf(r, __shfl_xor(r, 8, 32));
      wmax[v] = r;
    }
    if (ln == 0) {
#pragma unroll
      for (int v = 0; v < 8; ++v) red_max[w][v + 8 * h] = wmax[v];
    }
    __syncthreads();  // also fences previous iteration's plds reads

    float mnew[8], scale[8], p[8];
#pragma unroll
    for (int v = 0; v < 8; ++v) {
      float mt = -1e30f;
#pragma unroll
      for (int ww = 0; ww < 8; ++ww) mt = fmaxf(mt, red_max[ww][v + 8 * h]);
      float mn = fmaxf(m_run[v], mt);
      mnew[v]  = mn;
      scale[v] = __expf(m_run[v] - mn);
      p[v]     = __expf(sg[v] - mn);          // masked entries -> 0
    }

    // ---- row sum of p over this wave's columns
    float wsum[8];
#pragma unroll
    for (int v = 0; v < 8; ++v) {
      float r = p[v];
      r += __shfl_xor(r, 1, 32);
      r += __shfl_xor(r, 2, 32);
      r += __shfl_xor(r, 4, 32);
      r += __shfl_xor(r, 8, 32);
      wsum[v] = r;
    }
    if (ln == 0) {
#pragma unroll
      for (int v = 0; v < 8; ++v) red_sum[w][v + 8 * h] = wsum[v];
    }
    // ---- publish p tile (bf16) to LDS
    {
      const int col = 16 * w + ln;
#pragma unroll
      for (int v = 0; v < 8; ++v)
        plds[(v + 8 * h) * 128 + col] = f2bf(p[v]);
    }
    __syncthreads();

#pragma unroll
    for (int v = 0; v < 8; ++v) {
      float lt = 0.0f;
#pragma unroll
      for (int ww = 0; ww < 8; ++ww) lt += red_sum[ww][v + 8 * h];
      l_run[v] = l_run[v] * scale[v] + lt;
      m_run[v] = mnew[v];
    }
#pragma unroll
    for (int n = 0; n < 4; ++n)
#pragma unroll
      for (int v = 0; v < 8; ++v)
        acc[n][v] *= scale[v];

    // ---- acc += p[16x128] @ CT[128 x 64] for this wave's D_OUT chunk (16 wmma)
    if (nb + 256 < NPAD)
      __builtin_prefetch(CTt + (size_t)(64 * w) * NPAD + nb + 128, 0, 1);
#pragma unroll
    for (int kk = 0; kk < 4; ++kk) {
      const unsigned short* pl = plds + ln * 128 + 32 * kk + 16 * h;
      bf16x16 a = frag_split(pl, pl + 8);
#pragma unroll
      for (int n = 0; n < 4; ++n) {
        const int dout = 64 * w + 16 * n + ln;
        const unsigned short* ctp = CTt + (size_t)dout * NPAD + nb + 32 * kk + 16 * h;
        bf16x16 b = frag_contig(ctp);
        acc[n] = wmma_bf16(a, b, acc[n]);
      }
    }
  }

  // ---- partial outputs
  if (w == 0 && ln == 0) {
#pragma unroll
    for (int v = 0; v < 8; ++v) {
      int grow = bi * 16 + v + 8 * h;
      m_part[split * B_ROWS + grow] = m_run[v];
      l_part[split * B_ROWS + grow] = l_run[v];
    }
  }
#pragma unroll
  for (int n = 0; n < 4; ++n) {
#pragma unroll
    for (int v = 0; v < 8; ++v) {
      int grow = bi * 16 + v + 8 * h;
      int dout = 64 * w + 16 * n + ln;
      acc_part[((size_t)split * B_ROWS + grow) * D_OUT + dout] = acc[n][v];
    }
  }
}

// ---------------------------------------------------------------- split merge + epilogue
__global__ void __launch_bounds__(256) k_combine(const float* __restrict__ proj,
                                                 const float* __restrict__ m_part,
                                                 const float* __restrict__ l_part,
                                                 const float* __restrict__ acc_part,
                                                 const float* __restrict__ alpha_p,
                                                 float* __restrict__ out) {
  int i = blockIdx.x * 256 + threadIdx.x;
  if (i >= B_ROWS * D_OUT) return;
  int row = i >> 9;
  int d   = i & 511;
  float mstar = -1e30f;
#pragma unroll
  for (int s = 0; s < NSPLIT; ++s) mstar = fmaxf(mstar, m_part[s * B_ROWS + row]);
  float L = 0.0f, A = 0.0f;
#pragma unroll
  for (int s = 0; s < NSPLIT; ++s) {
    float e = __expf(m_part[s * B_ROWS + row] - mstar);
    L += l_part[s * B_ROWS + row] * e;
    A += acc_part[((size_t)s * B_ROWS + row) * D_OUT + d] * e;
  }
  out[i] = proj[i] + alpha_p[0] * (A / L);
}

// ---------------------------------------------------------------- host launch
extern "C" void kernel_launch(void* const* d_in, const int* in_sizes, int n_in,
                              void* d_out, int out_size, void* d_ws, size_t ws_size,
                              hipStream_t stream) {
  const float* x     = (const float*)d_in[0];
  const float* W     = (const float*)d_in[1];
  const float* b     = (const float*)d_in[2];
  const float* CI    = (const float*)d_in[3];
  const float* CT    = (const float*)d_in[4];
  const float* beta  = (const float*)d_in[5];
  const float* alpha = (const float*)d_in[6];
  float* out = (float*)d_out;
  (void)in_sizes; (void)n_in; (void)out_size; (void)ws_size;

  char* base = (char*)d_ws;
  size_t off = 0;
  auto take = [&](size_t bytes) -> void* {
    off = (off + 255) & ~(size_t)255;
    void* p = base + off;
    off += bytes;
    return p;
  };
  unsigned short* Xb   = (unsigned short*)take((size_t)B_ROWS * D_IN * 2);
  unsigned short* Wb   = (unsigned short*)take((size_t)D_OUT * D_IN * 2);
  unsigned short* CIb  = (unsigned short*)take((size_t)NPAD * D_OUT * 2);
  unsigned short* CTt  = (unsigned short*)take((size_t)D_OUT * NPAD * 2);
  float*          proj = (float*)take((size_t)B_ROWS * D_OUT * 4);
  unsigned short* Pb   = (unsigned short*)take((size_t)B_ROWS * D_OUT * 2);
  float*          m_p  = (float*)take((size_t)NSPLIT * B_ROWS * 4);
  float*          l_p  = (float*)take((size_t)NSPLIT * B_ROWS * 4);
  float*          a_p  = (float*)take((size_t)NSPLIT * B_ROWS * D_OUT * 4);

  {
    int n = B_ROWS * D_IN;
    k_cvt<<<(n + 255) / 256, 256, 0, stream>>>(x, Xb, n);
  }
  {
    int n = D_OUT * D_IN;
    k_cvt<<<(n + 255) / 256, 256, 0, stream>>>(W, Wb, n);
  }
  {
    int n = NPAD * D_OUT;
    k_cvt_pad_rows<<<(n + 255) / 256, 256, 0, stream>>>(CI, CIb, N_CACHE, NPAD, D_OUT);
  }
  {
    dim3 blk(32, 8), grd(NPAD / 32, D_OUT / 32);
    k_transpose_ct<<<grd, blk, 0, stream>>>(CT, CTt);
  }
  k_proj<<<(B_ROWS / 16) * (D_OUT / 16) / 8, 256, 0, stream>>>(Xb, Wb, b, proj, Pb);
  k_flash<<<(B_ROWS / 16) * NSPLIT, 256, 0, stream>>>(Pb, CIb, CTt, beta, m_p, l_p, a_p);
  k_combine<<<(B_ROWS * D_OUT + 255) / 256, 256, 0, stream>>>(proj, m_p, l_p, a_p, alpha, out);
}
